// GCNNet_59227599011891
// MI455X (gfx1250) — compile-verified, matrix operand order checked by reading, not run on hardware
//
#include <hip/hip_runtime.h>

// ---------------------------------------------------------------------------
// CDNA5 (gfx1250, wave32) GCN pipeline.
//  - GEMMs: v_wmma_f32_16x16x32_bf16, fp32 accumulate.
//  - Weights pre-packed to bf16 in WMMA fragment order (B frag = one 32B load).
//  - A fragments loaded as float4 (b128) with fused bias+ReLU, cvt on the fly.
//  - Edge aggregation: global f32 atomics (feature rows resident in 192MB L2).
// ---------------------------------------------------------------------------

typedef __attribute__((ext_vector_type(16))) __bf16 v16bf;
typedef __attribute__((ext_vector_type(8)))  float  v8f;

// K-index mapping for 16-bit WMMA A/B fragments (CDNA5 ISA 7.12.2):
// halves 0..7 -> K = hi*8 + h ; halves 8..15 -> K = 16 + hi*8 + (h-8)
__device__ __forceinline__ int wmma_kmap(int h, int hi) {
  return ((h & 8) << 1) + hi * 8 + (h & 7);
}

// Pack fp32 weight matrix B[K,N] (row-major) into bf16 fragment order:
// P[((kt*N + col)*2 + hi)*16 + h] = bf16(B[(kt*32 + kmap(h,hi))*N + col])
__global__ __launch_bounds__(256) void k_pack_b(const float* __restrict__ B,
                                                __bf16* __restrict__ P,
                                                int K, int N) {
  int t = blockIdx.x * blockDim.x + threadIdx.x;
  if (t >= K * N) return;
  int h   = t & 15;
  int hi  = (t >> 4) & 1;
  int cK  = t >> 5;             // kt*N + col
  int col = cK % N;
  int kt  = cK / N;
  int k   = kt * 32 + wmma_kmap(h, hi);
  P[t] = (__bf16)B[(size_t)k * N + col];
}

// D = act_out( act_in(A + bias_in) @ B + bias_out )
// A: [M,K] f32 row-major; Bp: packed bf16 fragments of B[K,N]; C: [M,N] f32.
// Requires N % 64 == 0, K % 32 == 0. One wave computes a 16x64 strip.
template<bool BIAS_IN, bool RELU_IN, bool BIAS_OUT, bool RELU_OUT>
__global__ __launch_bounds__(256) void gemm_bf16_wmma(
    const float* __restrict__ A, const __bf16* __restrict__ Bp,
    const float* __restrict__ bias_in, const float* __restrict__ bias_out,
    float* __restrict__ C, int M, int K, int N)
{
  const int lane = threadIdx.x & 31;          // wave32
  const int l    = lane & 15;
  const int hi   = lane >> 4;
  const int nStrips = N >> 6;
  const int mTiles  = (M + 15) >> 4;
  const int w = blockIdx.x * 8 + (threadIdx.x >> 5);
  if (w >= mTiles * nStrips) return;
  const int tm    = w / nStrips;
  const int strip = w - tm * nStrips;

  int row = tm * 16 + l;                      // A row supplied by this lane
  if (row >= M) row = M - 1;                  // clamp (duplicate loads are benign)
  const float* __restrict__ arow = A + (size_t)row * K;

  v8f acc[4];
  #pragma unroll
  for (int nt = 0; nt < 4; ++nt)
    #pragma unroll
    for (int r = 0; r < 8; ++r) acc[nt][r] = 0.0f;

  for (int k0 = 0; k0 < K; k0 += 32) {
    if (k0 + 32 < K) __builtin_prefetch(arow + k0 + 32, 0, 3);  // global_prefetch_b8

    // ---- A fragment: two contiguous 8-float runs per lane, b128 loads ----
    const float4* ap0 = (const float4*)(arow + k0 + hi * 8);        // halves 0..7
    const float4* ap1 = (const float4*)(arow + k0 + 16 + hi * 8);   // halves 8..15
    float4 u[4] = { ap0[0], ap0[1], ap1[0], ap1[1] };
    if (BIAS_IN) {
      const float4* bp0 = (const float4*)(bias_in + k0 + hi * 8);
      const float4* bp1 = (const float4*)(bias_in + k0 + 16 + hi * 8);
      float4 bv[4] = { bp0[0], bp0[1], bp1[0], bp1[1] };
      #pragma unroll
      for (int q = 0; q < 4; ++q) {
        u[q].x += bv[q].x; u[q].y += bv[q].y; u[q].z += bv[q].z; u[q].w += bv[q].w;
      }
    }
    v16bf a;
    #pragma unroll
    for (int q = 0; q < 4; ++q) {
      float e0 = u[q].x, e1 = u[q].y, e2 = u[q].z, e3 = u[q].w;
      if (RELU_IN) {
        e0 = fmaxf(e0, 0.0f); e1 = fmaxf(e1, 0.0f);
        e2 = fmaxf(e2, 0.0f); e3 = fmaxf(e3, 0.0f);
      }
      a[q * 4 + 0] = (__bf16)e0; a[q * 4 + 1] = (__bf16)e1;
      a[q * 4 + 2] = (__bf16)e2; a[q * 4 + 3] = (__bf16)e3;
    }

    const size_t ktBase = (size_t)(k0 >> 5) * N;
    #pragma unroll
    for (int nt = 0; nt < 4; ++nt) {
      const int col = (strip << 6) + (nt << 4) + l;   // B column held by this lane
      // ---- B fragment: one 32-byte contiguous load from packed buffer ----
      const v16bf b = *(const v16bf*)(Bp + (((ktBase + col) << 1) + hi) * 16);
      acc[nt] = __builtin_amdgcn_wmma_f32_16x16x32_bf16(
          /*neg_a=*/false, a, /*neg_b=*/false, b,
          /*c_mod=*/(short)0, acc[nt], /*reuse_a=*/false, /*reuse_b=*/false);
    }
  }

  #pragma unroll
  for (int nt = 0; nt < 4; ++nt) {
    const int col = (strip << 6) + (nt << 4) + l;
    float bo = BIAS_OUT ? bias_out[col] : 0.0f;
    #pragma unroll
    for (int r = 0; r < 8; ++r) {
      const int orow = tm * 16 + hi * 8 + r;          // C layout: VGPR r -> M = hi*8+r
      if (orow < M) {
        float v = acc[nt][r] + bo;
        if (RELU_OUT) v = fmaxf(v, 0.0f);
        C[(size_t)orow * N + col] = v;
      }
    }
  }
}

// ---------------- elementwise / scatter kernels ----------------

__global__ __launch_bounds__(256) void k_fill(float* __restrict__ p, float v, int n) {
  int i = blockIdx.x * blockDim.x + threadIdx.x;
  if (i < n) p[i] = v;
}

__global__ __launch_bounds__(256) void k_deg_accum(const int* __restrict__ dst,
                                                   float* __restrict__ deg, int e) {
  int i = blockIdx.x * blockDim.x + threadIdx.x;
  if (i < e) atomicAdd(&deg[dst[i]], 1.0f);
}

__global__ __launch_bounds__(256) void k_rsqrt(const float* __restrict__ deg,
                                               float* __restrict__ dinv, int n) {
  int i = blockIdx.x * blockDim.x + threadIdx.x;
  if (i < n) dinv[i] = rsqrtf(deg[i]);            // deg >= 1 always (self-loop)
}

// out[i][f] = h[i][f] * dinv[i]^2   (self-loop term; also initializes out)
__global__ __launch_bounds__(256) void k_selfloop(const float* __restrict__ h,
                                                  const float* __restrict__ dinv,
                                                  float* __restrict__ out,
                                                  int n, int fshift) {
  int t = blockIdx.x * blockDim.x + threadIdx.x;
  if (t < (n << fshift)) {
    int i = t >> fshift;
    float d = dinv[i];
    out[t] = h[t] * d * d;
  }
}

// out[dst][f..f+3] += h[src][f..f+3] * dinv[src]*dinv[dst]; 4 features/thread
__global__ __launch_bounds__(256) void k_edge_scatter(const float* __restrict__ h,
                                                      const int* __restrict__ src,
                                                      const int* __restrict__ dst,
                                                      const float* __restrict__ dinv,
                                                      float* __restrict__ out,
                                                      int e, int fshift /* log2(F/4) */) {
  int t = blockIdx.x * blockDim.x + threadIdx.x;
  if (t >= (e << fshift)) return;
  int ei = t >> fshift;
  int f4 = (t & ((1 << fshift) - 1)) << 2;
  int s = src[ei], d = dst[ei];
  float c = dinv[s] * dinv[d];
  const float4 hv = *(const float4*)(h + (((size_t)s << fshift) << 2) + f4);
  float* o = out + (((size_t)d << fshift) << 2) + f4;
  atomicAdd(o + 0, hv.x * c);
  atomicAdd(o + 1, hv.y * c);
  atomicAdd(o + 2, hv.z * c);
  atomicAdd(o + 3, hv.w * c);
}

__global__ __launch_bounds__(256) void k_pool_count(const int* __restrict__ batch,
                                                    float* __restrict__ cnt, int n) {
  int i = blockIdx.x * blockDim.x + threadIdx.x;
  if (i < n) atomicAdd(&cnt[batch[i]], 1.0f);
}

// gsum[batch[i]][f] += relu(h[i][f] + bias[f])
__global__ __launch_bounds__(256) void k_pool_sum(const float* __restrict__ h,
                                                  const float* __restrict__ bias,
                                                  const int* __restrict__ batch,
                                                  float* __restrict__ gsum,
                                                  int n, int fshift) {
  int t = blockIdx.x * blockDim.x + threadIdx.x;
  if (t >= (n << fshift)) return;
  int i = t >> fshift;
  int f = t & ((1 << fshift) - 1);
  float v = fmaxf(h[t] + bias[f], 0.0f);
  atomicAdd(&gsum[((size_t)batch[i] << fshift) + f], v);
}

__global__ __launch_bounds__(256) void k_pool_div(float* __restrict__ g,
                                                  const float* __restrict__ cnt,
                                                  int G, int fshift) {
  int t = blockIdx.x * blockDim.x + threadIdx.x;
  if (t < (G << fshift)) g[t] /= fmaxf(cnt[t >> fshift], 1.0f);
}

// out[g] = dot(e[g][0:K], Wo) + bo   (OUT = 1)
__global__ __launch_bounds__(256) void k_final(const float* __restrict__ e,
                                               const float* __restrict__ Wo,
                                               const float* __restrict__ bo,
                                               float* __restrict__ out, int G, int K) {
  int g = blockIdx.x * blockDim.x + threadIdx.x;
  if (g >= G) return;
  float s = 0.0f;
  for (int k = 0; k < K; ++k) s += e[(size_t)g * K + k] * Wo[k];
  out[g] = s + bo[0];
}

// ---------------- driver ----------------

static inline int ceil_div(long long a, int b) { return (int)((a + b - 1) / b); }

extern "C" void kernel_launch(void* const* d_in, const int* in_sizes, int n_in,
                              void* d_out, int out_size, void* d_ws, size_t ws_size,
                              hipStream_t stream) {
  const float* x     = (const float*)d_in[0];
  const int*   src   = (const int*)  d_in[1];
  const int*   dst   = (const int*)  d_in[2];
  const int*   batch = (const int*)  d_in[3];
  const float* Wc0 = (const float*)d_in[4];  const float* bc0 = (const float*)d_in[5];
  const float* Wc1 = (const float*)d_in[6];  const float* bc1 = (const float*)d_in[7];
  const float* Wf  = (const float*)d_in[8];  const float* bfv = (const float*)d_in[9];
  const float* Wh0 = (const float*)d_in[10]; const float* bh0 = (const float*)d_in[11];
  const float* Wh1 = (const float*)d_in[12]; const float* bh1 = (const float*)d_in[13];
  const float* Wh2 = (const float*)d_in[14]; const float* bh2 = (const float*)d_in[15];
  const float* Wo  = (const float*)d_in[16]; const float* bo  = (const float*)d_in[17];
  float* out = (float*)d_out;

  const int Nn = in_sizes[3];                 // nodes (100000)
  const int Ee = in_sizes[1];                 // edges (600000)
  const int Gg = out_size;                    // graphs (512, OUT==1)
  const int F_IN = in_sizes[0] / Nn;          // 128
  const int H0 = in_sizes[5];                 // 64
  const int H1 = in_sizes[7];                 // 128
  const int Dd = in_sizes[9];                 // 512
  const int s0 = __builtin_ctz(H0);
  const int s1 = __builtin_ctz(H1);

  // Workspace layout (floats). agg2 overlays the dead h1+agg1 region (N*128).
  float* ws = (float*)d_ws;
  size_t off = 0;
  float* deg  = ws + off; off += (size_t)Nn;
  float* dinv = ws + off; off += (size_t)Nn;
  float* h1   = ws + off; off += (size_t)Nn * H0;
  float* agg1 = ws + off; off += (size_t)Nn * H0;
  float* h2   = ws + off; off += (size_t)Nn * H1;
  float* agg2 = h1;                                  // reuse: Nn*(H0+H0) == Nn*H1
  float* cnt  = ws + off; off += (size_t)Gg;
  float* gsum = ws + off; off += (size_t)Gg * H1;
  float* e0   = ws + off; off += (size_t)Gg * Dd;
  float* e1   = ws + off; off += (size_t)Gg * (Dd / 2);
  float* e2   = ws + off; off += (size_t)Gg * (Dd / 4);
  float* e3   = ws + off; off += (size_t)Gg * (Dd / 8);
  // packed bf16 weights (K*N halves == K*N/2 floats each; all multiples of 16)
  __bf16* pWc0 = (__bf16*)(ws + off); off += (size_t)F_IN * H0 / 2;
  __bf16* pWc1 = (__bf16*)(ws + off); off += (size_t)H0 * H1 / 2;
  __bf16* pWf  = (__bf16*)(ws + off); off += (size_t)H1 * Dd / 2;
  __bf16* pWh0 = (__bf16*)(ws + off); off += (size_t)Dd * (Dd / 2) / 2;
  __bf16* pWh1 = (__bf16*)(ws + off); off += (size_t)(Dd / 2) * (Dd / 4) / 2;
  __bf16* pWh2 = (__bf16*)(ws + off); off += (size_t)(Dd / 4) * (Dd / 8) / 2;
  (void)ws_size;

  const int T = 256;
  auto gemm_blocks = [](int M, int N) { return (((M + 15) / 16) * (N / 64) + 7) / 8; };
  auto pack_blocks = [](int K, int N) { return (K * N + 255) / 256; };

  // ---- pack weights to bf16 fragment order (tiny; every call, replay-safe) ----
  k_pack_b<<<pack_blocks(F_IN, H0), T, 0, stream>>>(Wc0, pWc0, F_IN, H0);
  k_pack_b<<<pack_blocks(H0, H1),   T, 0, stream>>>(Wc1, pWc1, H0, H1);
  k_pack_b<<<pack_blocks(H1, Dd),   T, 0, stream>>>(Wf,  pWf,  H1, Dd);
  k_pack_b<<<pack_blocks(Dd, Dd/2), T, 0, stream>>>(Wh0, pWh0, Dd, Dd/2);
  k_pack_b<<<pack_blocks(Dd/2, Dd/4), T, 0, stream>>>(Wh1, pWh1, Dd/2, Dd/4);
  k_pack_b<<<pack_blocks(Dd/4, Dd/8), T, 0, stream>>>(Wh2, pWh2, Dd/4, Dd/8);

  // ---- degree normalization: deg = 1 + in-degree, dinv = rsqrt(deg) ----
  k_fill<<<ceil_div(Nn, T), T, 0, stream>>>(deg, 1.0f, Nn);
  k_deg_accum<<<ceil_div(Ee, T), T, 0, stream>>>(dst, deg, Ee);
  k_rsqrt<<<ceil_div(Nn, T), T, 0, stream>>>(deg, dinv, Nn);

  // ---- GCNConv 1: h1 = x @ Wc0 ; agg1 = sym-norm aggregate(h1) ----
  gemm_bf16_wmma<false,false,false,false><<<gemm_blocks(Nn, H0), T, 0, stream>>>(
      x, pWc0, nullptr, nullptr, h1, Nn, F_IN, H0);
  k_selfloop<<<ceil_div((long long)Nn << s0, T), T, 0, stream>>>(h1, dinv, agg1, Nn, s0);
  k_edge_scatter<<<ceil_div((long long)Ee << (s0 - 2), T), T, 0, stream>>>(
      h1, src, dst, dinv, agg1, Ee, s0 - 2);

  // ---- GCNConv 2: h2 = relu(agg1 + bc0) @ Wc1 (bias+ReLU fused into A load) ----
  gemm_bf16_wmma<true,true,false,false><<<gemm_blocks(Nn, H1), T, 0, stream>>>(
      agg1, pWc1, bc0, nullptr, h2, Nn, H0, H1);
  k_selfloop<<<ceil_div((long long)Nn << s1, T), T, 0, stream>>>(h2, dinv, agg2, Nn, s1);
  k_edge_scatter<<<ceil_div((long long)Ee << (s1 - 2), T), T, 0, stream>>>(
      h2, src, dst, dinv, agg2, Ee, s1 - 2);

  // ---- global mean pool (relu(agg2 + bc1) fused into the sum) ----
  k_fill<<<ceil_div(Gg * (1 + H1), T), T, 0, stream>>>(cnt, 0.0f, Gg * (1 + H1)); // cnt+gsum contiguous
  k_pool_count<<<ceil_div(Nn, T), T, 0, stream>>>(batch, cnt, Nn);
  k_pool_sum<<<ceil_div((long long)Nn << s1, T), T, 0, stream>>>(agg2, bc1, batch, gsum, Nn, s1);
  k_pool_div<<<ceil_div((long long)Gg << s1, T), T, 0, stream>>>(gsum, cnt, Gg, s1);

  // ---- MLP head ----
  gemm_bf16_wmma<false,false,true,false><<<gemm_blocks(Gg, Dd), T, 0, stream>>>(
      gsum, pWf, nullptr, bfv, e0, Gg, H1, Dd);
  gemm_bf16_wmma<false,false,true,true><<<gemm_blocks(Gg, Dd/2), T, 0, stream>>>(
      e0, pWh0, nullptr, bh0, e1, Gg, Dd, Dd/2);
  gemm_bf16_wmma<false,false,true,true><<<gemm_blocks(Gg, Dd/4), T, 0, stream>>>(
      e1, pWh1, nullptr, bh1, e2, Gg, Dd/2, Dd/4);
  gemm_bf16_wmma<false,false,true,true><<<gemm_blocks(Gg, Dd/8), T, 0, stream>>>(
      e2, pWh2, nullptr, bh2, e3, Gg, Dd/4, Dd/8);
  k_final<<<ceil_div(Gg, T), T, 0, stream>>>(e3, Wo, bo, out, Gg, Dd/8);
}
